// MultiHeadAttention_88261577933125
// MI455X (gfx1250) — matrix operand order repaired and checked by reading, not compiled, not run on hardware
//
#include <hip/hip_runtime.h>
#include <hip/hip_bf16.h>

// MI455X / gfx1250, wave32. All GEMMs on v_wmma_f32_16x16x32_f16.
// Pipeline: QKV projection GEMMs -> RoPE -> flash attention -> output GEMM.
// Flash attention stages K/V tiles into LDS with global_load_async_to_lds_b128
// (ASYNCcnt-tracked), double-buffered across iterations.

typedef __attribute__((ext_vector_type(16))) _Float16 v16h;
typedef __attribute__((ext_vector_type(8)))  _Float16 v8h;
typedef __attribute__((ext_vector_type(4)))  _Float16 v4h;
typedef __attribute__((ext_vector_type(8)))  float    v8f;

#define CHN 1024
#define TT  2048
#define NH  16
#define KC  64
#define NB  4

static __device__ __forceinline__ v8f wmma_f16(v16h a, v16h b, v8f c) {
  // (neg_a, A, neg_b, B, c_mod, C, reuse_a, reuse_b)
  return __builtin_amdgcn_wmma_f32_16x16x32_f16(false, a, false, b, (short)0, c,
                                                false, false);
}

// Async global -> LDS copy of one 16B chunk per lane (tracked by ASYNCcnt).
static __device__ __forceinline__ void async_g2l_b128(unsigned lds_off,
                                                      const void* gaddr) {
  asm volatile("global_load_async_to_lds_b128 %0, %1, off"
               :: "v"(lds_off), "v"(gaddr) : "memory");
}
static __device__ __forceinline__ void wait_asynccnt0() {
  asm volatile("s_wait_asynccnt 0x0" ::: "memory");
}

// ---------------------------------------------------------------------------
// Generic WMMA GEMM:  out[b,o,t] = sum_i W[o,i] * X[b,i,t] + bias[o]
// OUT_MODE 0: f16 out at [b][h][t][c]   (q/k layout, c contiguous)
// OUT_MODE 1: f16 out at [b][o][t]      (v layout == [b][h][c][t], t contiguous)
// OUT_MODE 2: f32 out at [b][o][t]      (final output)
// Block = 256 threads (8 waves); tile 128(M=o) x 128(N=t); K-panel = 32.
// ---------------------------------------------------------------------------
template <typename InT, int OUT_MODE>
__global__ __launch_bounds__(256)
void gemm_wmma(const float* __restrict__ W, const InT* __restrict__ X,
               const float* __restrict__ bias, void* __restrict__ outp)
{
  __shared__ alignas(32) _Float16 lA[128 * 40];  // [m][k], row stride 40 halfs (80B)
  __shared__ alignas(32) _Float16 lB[128 * 48];  // [t][k], row stride 48 halfs (96B)

  const int tid = threadIdx.x;
  const int b   = blockIdx.z;
  const int o0  = blockIdx.y * 128;
  const int t0  = blockIdx.x * 128;

  const int w    = tid >> 5;
  const int lane = tid & 31;
  const int hlf  = lane >> 4;   // which 16-lane half
  const int l16  = lane & 15;
  const int wm   = (w & 3) * 32;   // wave M offset within 128
  const int wn   = (w >> 2) * 64;  // wave N offset within 128

  v8f acc[2][4];
  #pragma unroll
  for (int a = 0; a < 2; ++a)
    #pragma unroll
    for (int nf = 0; nf < 4; ++nf) acc[a][nf] = v8f{};

  for (int kk = 0; kk < CHN; kk += 32) {
    __syncthreads();
    // --- stage A panel: W[o0..o0+127][kk..kk+31] -> lA[m][k] (f16) ---
    #pragma unroll
    for (int it = 0; it < 4; ++it) {
      int id  = it * 256 + tid;          // 1024 float4 loads
      int row = id >> 3;
      int c4  = (id & 7) * 4;
      const float4 w4 = *(const float4*)(W + (size_t)(o0 + row) * CHN + kk + c4);
      _Float16* dst = &lA[row * 40 + c4];
      dst[0] = (_Float16)w4.x; dst[1] = (_Float16)w4.y;
      dst[2] = (_Float16)w4.z; dst[3] = (_Float16)w4.w;
    }
    // --- stage B panel: X[b][kk..kk+31][t0..t0+127] -> lB[t][k] (f16, transposed) ---
    #pragma unroll
    for (int it = 0; it < 4; ++it) {
      int id = it * 256 + tid;           // 1024 4-element loads
      int kr = id >> 5;
      int c4 = (id & 31) * 4;
      const InT* src = X + (size_t)(b * CHN + kk + kr) * TT + t0 + c4;
      _Float16 h0, h1, h2, h3;
      if constexpr (sizeof(InT) == 4) {
        float4 f = *(const float4*)src;
        h0 = (_Float16)f.x; h1 = (_Float16)f.y; h2 = (_Float16)f.z; h3 = (_Float16)f.w;
      } else {
        v4h f = *(const v4h*)src;
        h0 = f[0]; h1 = f[1]; h2 = f[2]; h3 = f[3];
      }
      lB[(c4 + 0) * 48 + kr] = h0;
      lB[(c4 + 1) * 48 + kr] = h1;
      lB[(c4 + 2) * 48 + kr] = h2;
      lB[(c4 + 3) * 48 + kr] = h3;
    }
    // prefetch next A panel into caches
    if (kk + 32 < CHN)
      __builtin_prefetch((const void*)(W + (size_t)(o0 + (tid >> 1)) * CHN + kk + 32 +
                                       (tid & 1) * 16), 0, 1);
    __syncthreads();

    // --- fragments + 8 WMMA per wave ---
    v16h af[2];
    #pragma unroll
    for (int a = 0; a < 2; ++a) {
      // A frag (16x32 f16): lane l16 = row m; contiguous K order, same as B:
      // k = j + 16*hlf  -> two contiguous 8-half (16B) runs.
      const _Float16* p = &lA[(wm + 16 * a + l16) * 40];
      v8h lo = *(const v8h*)(p + 16 * hlf);
      v8h hi = *(const v8h*)(p + 16 * hlf + 8);
      v16h f;
      #pragma unroll
      for (int j = 0; j < 8; ++j) { f[j] = lo[j]; f[j + 8] = hi[j]; }
      af[a] = f;
    }
    #pragma unroll
    for (int nf = 0; nf < 4; ++nf) {
      // B frag (32x16 f16): lane l16 = col n; k = j + 16*hlf -> one 32B run.
      v16h bf = *(const v16h*)(&lB[(wn + 16 * nf + l16) * 48 + 16 * hlf]);
      #pragma unroll
      for (int a = 0; a < 2; ++a) acc[a][nf] = wmma_f16(af[a], bf, acc[a][nf]);
    }
  }

  // --- epilogue: D layout is (M = r + 8*hlf, N = l16) per vgpr r ---
  #pragma unroll
  for (int a = 0; a < 2; ++a) {
    #pragma unroll
    for (int nf = 0; nf < 4; ++nf) {
      const int obase = o0 + wm + 16 * a + 8 * hlf;  // row for r=0 (mult of 8)
      const int tcol  = t0 + wn + 16 * nf + l16;
      if constexpr (OUT_MODE == 0) {
        _Float16* out = (_Float16*)outp;
        const int hh = obase >> 6, c0 = obase & 63;   // never crosses a head
        v8h vals;
        #pragma unroll
        for (int r = 0; r < 8; ++r)
          vals[r] = (_Float16)(acc[a][nf][r] + bias[obase + r]);
        *(v8h*)&out[(((size_t)b * NH + hh) * TT + tcol) * KC + c0] = vals;
      } else if constexpr (OUT_MODE == 1) {
        _Float16* out = (_Float16*)outp;
        #pragma unroll
        for (int r = 0; r < 8; ++r) {
          const int o = obase + r;
          out[((size_t)b * CHN + o) * TT + tcol] =
              (_Float16)(acc[a][nf][r] + bias[o]);
        }
      } else {
        float* out = (float*)outp;
        #pragma unroll
        for (int r = 0; r < 8; ++r) {
          const int o = obase + r;
          out[((size_t)b * CHN + o) * TT + tcol] = acc[a][nf][r] + bias[o];
        }
      }
    }
  }
}

// ---------------------------------------------------------------------------
// RoPE (d = 32, d/2 = 16) applied in place to q,k stored f16 [b][h][t][64].
// ---------------------------------------------------------------------------
__global__ __launch_bounds__(256)
void rope_kernel(_Float16* __restrict__ q, _Float16* __restrict__ k)
{
  const int idx = blockIdx.x * 256 + threadIdx.x;   // over B*H*T*16 pairs
  const int j  = idx & 15;
  const int t  = (idx >> 4) & (TT - 1);
  const int bh = idx >> 15;
  const float theta = __powf(10000.f, -(float)j * (1.f / 16.f));
  float s, c;
  __sincosf((float)t * theta, &s, &c);
  const size_t base = ((size_t)bh * TT + t) * KC;
  {
    float x1 = (float)q[base + j], x2 = (float)q[base + j + 16];
    q[base + j]      = (_Float16)(x1 * c - x2 * s);
    q[base + j + 16] = (_Float16)(x2 * c + x1 * s);
  }
  {
    float x1 = (float)k[base + j], x2 = (float)k[base + j + 16];
    k[base + j]      = (_Float16)(x1 * c - x2 * s);
    k[base + j + 16] = (_Float16)(x2 * c + x1 * s);
  }
}

// ---------------------------------------------------------------------------
// Flash attention: grid (T/128, B*H), block 256 (8 waves). Each wave owns 16
// query rows; K/V streamed in 64-wide tiles, double-buffered in LDS and filled
// by async global->LDS copies so every wave reads each tile from LDS instead
// of 8x-redundant global loads. Online softmax, S and P*V on WMMA; P is
// re-striped D-layout -> A-layout through wave-private LDS.
// q,k: f16 [b][h][t][64]; vt: f16 [b][h][c][t]; out: f16 [b][C][T].
// ---------------------------------------------------------------------------
__global__ __launch_bounds__(256)
void flash_attn(const _Float16* __restrict__ q, const _Float16* __restrict__ k,
                const _Float16* __restrict__ vt, const int* __restrict__ mask,
                _Float16* __restrict__ out)
{
  // Row stride 80 halfs (160B, multiple of 32B) keeps v16h reads aligned.
  __shared__ alignas(32) _Float16 lK[2][64 * 80];  // [t][c] tile, 10KB each
  __shared__ alignas(32) _Float16 lV[2][64 * 80];  // [c][t] tile, 10KB each
  __shared__ alignas(32) _Float16 lP[8][16 * 72];  // per-wave P, stride 72

  const int tid  = threadIdx.x;
  const int w    = tid >> 5;
  const int lane = tid & 31;
  const int hlf  = lane >> 4;
  const int l16  = lane & 15;
  const int bh   = blockIdx.y;
  const int b    = bh >> 4, h = bh & 15;
  const int q0   = blockIdx.x * 128 + w * 16;      // first query row of wave

  // Issue async copies for one K/V tile (64 rows x 64 halfs each, 16B chunks).
  auto stage = [&](int buf, int kt) {
    #pragma unroll
    for (int it = 0; it < 2; ++it) {
      int id  = it * 256 + tid;                 // 512 chunks of 16B
      int row = id >> 3, c8 = (id & 7) * 8;
      const _Float16* g = k + ((size_t)bh * TT + kt + row) * KC + c8;
      async_g2l_b128((unsigned)(uintptr_t)&lK[buf][row * 80 + c8], g);
    }
    #pragma unroll
    for (int it = 0; it < 2; ++it) {
      int id = it * 256 + tid;
      int cr = id >> 3, t8 = (id & 7) * 8;
      const _Float16* g = vt + ((size_t)bh * KC + cr) * TT + kt + t8;
      async_g2l_b128((unsigned)(uintptr_t)&lV[buf][cr * 80 + t8], g);
    }
  };

  // Q fragments (two K=32 chunks covering c=0..63), loaded once.
  const _Float16* qrow = q + ((size_t)bh * TT + q0 + l16) * KC;
  v16h aq[2];
  #pragma unroll
  for (int ci = 0; ci < 2; ++ci) {
    const int c0 = ci * 32;
    v8h lo = *(const v8h*)(qrow + c0 + 16 * hlf);
    v8h hi = *(const v8h*)(qrow + c0 + 16 * hlf + 8);
    v16h f;
    #pragma unroll
    for (int j = 0; j < 8; ++j) { f[j] = lo[j]; f[j + 8] = hi[j]; }
    aq[ci] = f;
  }

  float mrow[8], lrow[8];
  #pragma unroll
  for (int r = 0; r < 8; ++r) { mrow[r] = -3.0e38f; lrow[r] = 0.f; }
  v8f oacc[4];
  #pragma unroll
  for (int cf = 0; cf < 4; ++cf) oacc[cf] = v8f{};

  _Float16* lp = &lP[w][0];

  stage(0, 0);                         // prologue: fill buffer 0

  for (int kt = 0; kt < TT; kt += 64) {
    const int buf = (kt >> 6) & 1;
    // Own async writes landed; barrier makes every wave's writes visible and
    // guarantees everyone finished reading buf^1 before we refill it.
    wait_asynccnt0();
    __syncthreads();
    if (kt + 64 < TT) stage(buf ^ 1, kt + 64);   // prefetch next tile

    // ---- S = (Q K^T) / 8 over this 16x64 tile (K frags from LDS) ----
    float sv[4][8];
    #pragma unroll
    for (int nf = 0; nf < 4; ++nf) {
      const _Float16* krow = &lK[buf][(nf * 16 + l16) * 80];
      v16h b0 = *(const v16h*)(krow + 16 * hlf);        // c = 0..31
      v16h b1 = *(const v16h*)(krow + 32 + 16 * hlf);   // c = 32..63
      v8f s = v8f{};
      s = wmma_f16(aq[0], b0, s);
      s = wmma_f16(aq[1], b1, s);
      #pragma unroll
      for (int r = 0; r < 8; ++r) sv[nf][r] = s[r] * 0.125f;
    }
    // ---- mask (reference: where mask==0 -> -10000) ----
    #pragma unroll
    for (int nf = 0; nf < 4; ++nf) {
      const int tk = kt + nf * 16 + l16;
      #pragma unroll
      for (int r = 0; r < 8; ++r) {
        const int qg = q0 + 8 * hlf + r;
        if (mask[(size_t)qg * TT + tk] == 0) sv[nf][r] = -10000.f;
      }
    }
    // ---- online softmax (rows live in one 16-lane half each) ----
    float osc[8];
    #pragma unroll
    for (int r = 0; r < 8; ++r) {
      float mx = fmaxf(fmaxf(sv[0][r], sv[1][r]), fmaxf(sv[2][r], sv[3][r]));
      #pragma unroll
      for (int d = 1; d < 16; d <<= 1) mx = fmaxf(mx, __shfl_xor(mx, d, 16));
      const float mn = fmaxf(mrow[r], mx);
      const float sc = __expf(mrow[r] - mn);
      float rs = 0.f;
      #pragma unroll
      for (int nf = 0; nf < 4; ++nf) {
        const float p = __expf(sv[nf][r] - mn);
        sv[nf][r] = p;
        rs += p;
      }
      #pragma unroll
      for (int d = 1; d < 16; d <<= 1) rs += __shfl_xor(rs, d, 16);
      lrow[r] = lrow[r] * sc + rs;
      mrow[r] = mn;
      osc[r]  = sc;
    }
    #pragma unroll
    for (int cf = 0; cf < 4; ++cf)
      #pragma unroll
      for (int r = 0; r < 8; ++r) oacc[cf][r] *= osc[r];

    // ---- P: D-layout regs -> wave-private LDS [m][k] -> A fragments ----
    #pragma unroll
    for (int nf = 0; nf < 4; ++nf)
      #pragma unroll
      for (int r = 0; r < 8; ++r)
        lp[(8 * hlf + r) * 72 + nf * 16 + l16] = (_Float16)sv[nf][r];

    v16h ap[2];
    #pragma unroll
    for (int ki = 0; ki < 2; ++ki) {
      const int kc = ki * 32;
      const _Float16* pr = lp + l16 * 72;
      v8h lo = *(const v8h*)(pr + kc + 16 * hlf);
      v8h hi = *(const v8h*)(pr + kc + 16 * hlf + 8);
      v16h f;
      #pragma unroll
      for (int j = 0; j < 8; ++j) { f[j] = lo[j]; f[j + 8] = hi[j]; }
      ap[ki] = f;
    }
    // ---- O += P * V  (V frags from LDS: row = channel, t contiguous) ----
    #pragma unroll
    for (int cf = 0; cf < 4; ++cf) {
      const _Float16* vrow = &lV[buf][(cf * 16 + l16) * 80];
      v16h b0 = *(const v16h*)(vrow + 16 * hlf);
      v16h b1 = *(const v16h*)(vrow + 32 + 16 * hlf);
      oacc[cf] = wmma_f16(ap[0], b0, oacc[cf]);
      oacc[cf] = wmma_f16(ap[1], b1, oacc[cf]);
    }
  }

  // ---- finalize: O / l, write f16 [b][C][T] (t contiguous over r) ----
  #pragma unroll
  for (int cf = 0; cf < 4; ++cf) {
    v8h vals;
    #pragma unroll
    for (int r = 0; r < 8; ++r)
      vals[r] = (_Float16)(oacc[cf][r] / fmaxf(lrow[r], 1e-20f));
    const int cg = h * KC + cf * 16 + l16;
    const size_t tbase = (size_t)q0 + 8 * hlf;
    *(v8h*)&out[((size_t)b * CHN + cg) * TT + tbase] = vals;
  }
}

// ---------------------------------------------------------------------------
extern "C" void kernel_launch(void* const* d_in, const int* in_sizes, int n_in,
                              void* d_out, int out_size, void* d_ws, size_t ws_size,
                              hipStream_t stream)
{
  const float* x  = (const float*)d_in[0];
  const float* c  = (const float*)d_in[1];
  const int* mask = (const int*)d_in[2];
  const float* wq = (const float*)d_in[3];
  const float* bq = (const float*)d_in[4];
  const float* wk = (const float*)d_in[5];
  const float* bk = (const float*)d_in[6];
  const float* wv = (const float*)d_in[7];
  const float* bv = (const float*)d_in[8];
  const float* wo = (const float*)d_in[9];
  const float* bo = (const float*)d_in[10];

  // Workspace: 4 f16 tensors of B*C*T = 8M elems (16MB each, 64MB total).
  const size_t NE = (size_t)NB * NH * TT * KC;
  _Float16* qb  = (_Float16*)d_ws;
  _Float16* kb  = qb + NE;
  _Float16* vb  = kb + NE;
  _Float16* att = vb + NE;

  dim3 gg(TT / 128, CHN / 128, NB), bb(256);
  gemm_wmma<float, 0><<<gg, bb, 0, stream>>>(wq, x, bq, (void*)qb);
  gemm_wmma<float, 0><<<gg, bb, 0, stream>>>(wk, c, bk, (void*)kb);
  gemm_wmma<float, 1><<<gg, bb, 0, stream>>>(wv, c, bv, (void*)vb);

  rope_kernel<<<(NB * NH * TT * 16) / 256, 256, 0, stream>>>(qb, kb);

  flash_attn<<<dim3(TT / 128, NB * NH), 256, 0, stream>>>(qb, kb, vb, mask, att);

  gemm_wmma<_Float16, 2><<<gg, bb, 0, stream>>>(wo, att, bo, d_out);
}
